// IndependentCNN_50989851738137
// MI455X (gfx1250) — matrix operand order repaired and checked by reading, not compiled, not run on hardware
//
#include <hip/hip_runtime.h>
#include <math.h>

typedef __attribute__((ext_vector_type(16))) _Float16 v16h;
typedef __attribute__((ext_vector_type(8)))  float    v8f;

#define NB   8192
#define NL   600
#define NC   30
#define NF   150        // features per channel
#define NH   128        // mlp hidden
#define NROW (NB*NC)    // 245760 GEMM rows
#define KPAD 160        // 150 padded to 5 * 32
#define NKS  5          // k-steps of 32
#define NNT  8          // n-tiles of 16 (128 cols)

// workspace byte offsets
#define OFF_S    0                    // 8192 f32
#define OFF_BP   32768                // 64 f32 block partials
#define OFF_P    33024                // 4500 f32
#define OFF_Q    51024                // 4500 f32
#define OFF_BSW  69024                // 20480 f16 (32B aligned)
#define OFF_KL   109984               // 3840 f32
#define OUT_ELEMS 36864000

// ---------------- Kernel A: s_b, moments, W1 -> swizzled f16 B fragments ---
__global__ void prep_a(const float* __restrict__ x, const float* __restrict__ w1,
                       float* __restrict__ s_ws, float* __restrict__ bparts,
                       _Float16* __restrict__ Bsw) {
  int t = blockIdx.x * 256 + threadIdx.x;           // 0..8191
  float s = 0.5f * (x[t * NL + 299] + x[t * NL + 300]);
  s_ws[t] = s;
  __shared__ float r1[256], r2[256];
  r1[threadIdx.x] = s; r2[threadIdx.x] = s * s;
  __syncthreads();
  for (int st = 128; st > 0; st >>= 1) {
    if (threadIdx.x < st) { r1[threadIdx.x] += r1[threadIdx.x + st];
                            r2[threadIdx.x] += r2[threadIdx.x + st]; }
    __syncthreads();
  }
  if (threadIdx.x == 0) { bparts[blockIdx.x*2] = r1[0]; bparts[blockIdx.x*2+1] = r2[0]; }

  // B fragment layout: per (ks,nt): 32 lanes x 16 halves.
  // lane = half*16 + N ; K = 16*half + 2*v + j  (B: contiguous 16-K per lane half)
  for (int i = t; i < NKS*NNT*32*16; i += NB) {
    int j  = i & 1;
    int v  = (i >> 1) & 7;
    int lt = (i >> 4) & 31;
    int rest = i >> 9;
    int nt = rest & 7, ks = rest >> 3;
    int N = lt & 15, half = lt >> 4;
    int K = 16*half + 2*v + j;
    int kg = ks*32 + K;
    int n  = nt*16 + N;
    float val = (kg < NF) ? w1[n*NF + kg] : 0.f;
    Bsw[i] = (_Float16)val;
  }
}

// ---------------- Kernel B: closed-form BN stats -> P/Q tables -------------
__global__ void prep_b(const float* __restrict__ fc_w, const float* __restrict__ fc_b,
                       const float* __restrict__ gam, const float* __restrict__ bet,
                       const float* __restrict__ bparts,
                       float* __restrict__ Pb, float* __restrict__ Qb) {
  __shared__ float Ws[NF];
  __shared__ float red[256];
  __shared__ float Sv[5];
  __shared__ float smu, srs;
  int t = threadIdx.x;
  float w = 0.f, cb = 0.f;
  if (t < NF) {
    float acc = 0.f;
    for (int d = 0; d < 20; ++d) acc += fc_w[t*20 + d];
    w = acc; cb = fc_b[t]; Ws[t] = acc;
  }
  float vals[5] = { w, w*w, w*cb, cb, cb*cb };
  for (int q = 0; q < 5; ++q) {
    red[t] = vals[q]; __syncthreads();
    for (int st = 128; st > 0; st >>= 1) { if (t < st) red[t] += red[t + st]; __syncthreads(); }
    if (t == 0) Sv[q] = red[0];
    __syncthreads();
  }
  if (t == 0) {
    float ss = 0.f, s2 = 0.f;
    for (int i = 0; i < 32; ++i) { ss += bparts[2*i]; s2 += bparts[2*i+1]; }
    float m1 = ss * (1.f/NB), m2 = s2 * (1.f/NB);
    float mu  = (m1*Sv[0] + Sv[3]) * (1.f/NF);
    float E2  = (m2*Sv[1] + 2.f*m1*Sv[2] + Sv[4]) * (1.f/NF);
    float var = E2 - mu*mu;
    smu = mu;
    srs = 1.f / sqrtf(var + 1e-5f);
  }
  __syncthreads();
  for (int i = t; i < NC*NF; i += 256) {
    int c = i / NF, f = i - c*NF;
    float g = gam[c];
    Pb[i] = Ws[f] * srs * g;
    Qb[i] = (fc_b[f] - smu) * srs * g + bet[c];
  }
}

// ---------------- Kernel C: main (WMMA GEMM + stream ib_x + KL) ------------
__global__ __launch_bounds__(128) void main_c(
    const float* __restrict__ s_ws, const float* __restrict__ Pb,
    const float* __restrict__ Qb,   const _Float16* __restrict__ Bsw,
    const float* __restrict__ b1,   const float* __restrict__ w2,
    const float* __restrict__ b2,   const float* __restrict__ noise,
    float* __restrict__ out,        float* __restrict__ klpart) {
  __shared__ float AshF[4][16*NF];                        // fp32 hn tiles
  __shared__ __align__(32) _Float16 Ash16[4][NKS*32*16];  // WMMA A fragments
  __shared__ float cmsh[4][16], cssh[4][16], impsh[4][16];
  __shared__ float rkc[4][16], rki[4][16], rsd[4][16];
  __shared__ float red[128];

  int tid  = threadIdx.x;
  int wave = tid >> 5, lane = tid & 31;
  int g0 = blockIdx.x * 64 + wave * 16;                   // first global row

  // --- fill hn tile: fp32 copy + f16 A-fragment (A: contiguous 16-K per lane half)
  for (int idx = lane; idx < 16*KPAD; idx += 32) {        // 2560 = 80*32, uniform
    int r = idx / KPAD, f = idx - r*KPAD;
    float val = 0.f;
    if (f < NF) {
      int g = g0 + r; int b = g / NC, c = g - b*NC;
      val = fmaxf(fmaf(s_ws[b], Pb[c*NF + f], Qb[c*NF + f]), 0.f);
      AshF[wave][r*NF + f] = val;
    }
    int ks = f >> 5, K = f & 31;
    int half = K >> 4;
    int v = (K & 15) >> 1;
    int j = K & 1;
    int lt = (half << 4) | r;
    Ash16[wave][((ks*32 + lt) << 4) + (v << 1) + j] = (_Float16)val;
  }
  __syncthreads();

  // --- per-row mean / std (ddof=1), two-pass for accuracy
  if (lane < 16) {
    const float2* row = (const float2*)&AshF[wave][lane*NF];
    float sum = 0.f;
    for (int k = 0; k < 75; ++k) { float2 p = row[k]; sum += p.x + p.y; }
    float mean = sum * (1.f/NF);
    float vs = 0.f;
    for (int k = 0; k < 75; ++k) {
      float2 p = row[k];
      float d0 = p.x - mean, d1 = p.y - mean;
      vs = fmaf(d0, d0, fmaf(d1, d1, vs));
    }
    cmsh[wave][lane] = mean;
    cssh[wave][lane] = sqrtf(vs * (1.f/149.f));
  }

  // --- z = hn * W1^T via v_wmma_f32_16x16x32_f16 (16x128 tile, K padded 160)
  v8f acc[NNT];
  #pragma unroll
  for (int nt = 0; nt < NNT; ++nt)
    #pragma unroll
    for (int e = 0; e < 8; ++e) acc[nt][e] = 0.f;

  const v16h* Ap = (const v16h*)&Ash16[wave][0];
  const v16h* Bp = (const v16h*)Bsw;
  #pragma unroll
  for (int ks = 0; ks < NKS; ++ks) {
    v16h Av = Ap[ks*32 + lane];
    #pragma unroll
    for (int nt = 0; nt < NNT; ++nt) {
      v16h Bv = Bp[(ks*NNT + nt)*32 + lane];
      acc[nt] = __builtin_amdgcn_wmma_f32_16x16x32_f16(
          false, Av, false, Bv, (short)0, acc[nt], false, false);
    }
  }

  // --- epilogue: relu(z+b1), dot with w2 across N, cross-lane reduce -> imp
  float dot[8];
  #pragma unroll
  for (int rr = 0; rr < 8; ++rr) dot[rr] = 0.f;
  int nl = lane & 15;
  #pragma unroll
  for (int nt = 0; nt < NNT; ++nt) {
    int h = nt*16 + nl;
    float b1v = b1[h], w2v = w2[h];
    #pragma unroll
    for (int rr = 0; rr < 8; ++rr) {
      float z = fmaxf(acc[nt][rr] + b1v, 0.f);
      dot[rr] = fmaf(z, w2v, dot[rr]);
    }
  }
  #pragma unroll
  for (int rr = 0; rr < 8; ++rr) {
    float d = dot[rr];
    d += __shfl_xor(d, 1, 32);
    d += __shfl_xor(d, 2, 32);
    d += __shfl_xor(d, 4, 32);
    d += __shfl_xor(d, 8, 32);     // stays within 16-lane halves
    dot[rr] = d;
  }
  float b2v = b2[0];
  #pragma unroll
  for (int rr = 0; rr < 8; ++rr) {
    if (nl == rr) {
      int m = ((lane >> 4) << 3) | rr;   // C/D layout: M = (lane/16)*8 + r
      impsh[wave][m] = 1.f / (1.f + expf(-(dot[rr] + b2v)));
    }
  }
  __syncthreads();

  // --- per-row KL constants (hoists all divides out of the streaming loop)
  if (lane < 16) {
    float im = impsh[wave][lane];
    float cs = cssh[wave][lane];
    float om = 1.f - im;
    float sd = om * cs;
    float ce = cs + 1e-8f, se = sd + 1e-8f;
    rkc[wave][lane] = 0.5f * ((sd*sd)/(ce*ce) + (cs*cs)/(se*se) - 1.f);
    rki[wave][lane] = 1.f / (ce*ce);
    rsd[wave][lane] = sd;
  }
  __syncthreads();

  // --- streaming output: ib_x (fp32 path) + KL accumulation
  float kls = 0.f;
  for (int idx = lane; idx < 16*NF; idx += 32) {          // 2400 = 75*32
    int r = idx / NF, f = idx - r*NF;
    int g = g0 + r;
    float a   = AshF[wave][r*NF + f];
    float im  = impsh[wave][r];
    float cmv = cmsh[wave][r];
    float om  = 1.f - im;
    float mean = fmaf(a, im, om * cmv);
    float sd   = rsd[wave][r];
    float nz = __builtin_nontemporal_load(&noise[g*NF + f]);
    __builtin_nontemporal_store(fmaf(nz, sd, mean), &out[g*NF + f]);
    float dm = mean - cmv;
    kls += rkc[wave][r] + dm*dm*rki[wave][r];
  }
  red[tid] = kls;
  __syncthreads();
  for (int st = 64; st > 0; st >>= 1) { if (tid < st) red[tid] += red[tid + st]; __syncthreads(); }
  if (tid == 0) klpart[blockIdx.x] = red[0];
}

// ---------------- Kernel D: deterministic KL mean --------------------------
__global__ void reduce_d(const float* __restrict__ klpart, float* __restrict__ out) {
  __shared__ float red[256];
  int t = threadIdx.x;
  float s = 0.f;
  for (int i = t; i < 3840; i += 256) s += klpart[i];
  red[t] = s; __syncthreads();
  for (int st = 128; st > 0; st >>= 1) { if (t < st) red[t] += red[t + st]; __syncthreads(); }
  if (t == 0) out[OUT_ELEMS] = red[0] * (1.0f / (float)OUT_ELEMS);
}

extern "C" void kernel_launch(void* const* d_in, const int* in_sizes, int n_in,
                              void* d_out, int out_size, void* d_ws, size_t ws_size,
                              hipStream_t stream) {
  const float* x     = (const float*)d_in[0];
  // d_in[1] offset_w, d_in[2] offset_b: provably dead (W=1 => ix=0 always)
  const float* fc_w  = (const float*)d_in[3];
  const float* fc_b  = (const float*)d_in[4];
  const float* gam   = (const float*)d_in[5];
  const float* bet   = (const float*)d_in[6];
  const float* w1    = (const float*)d_in[7];
  const float* b1    = (const float*)d_in[8];
  const float* w2    = (const float*)d_in[9];
  const float* b2    = (const float*)d_in[10];
  const float* noise = (const float*)d_in[11];
  float* out = (float*)d_out;
  char*  ws  = (char*)d_ws;

  float*     s_ws   = (float*)(ws + OFF_S);
  float*     bparts = (float*)(ws + OFF_BP);
  float*     Pb     = (float*)(ws + OFF_P);
  float*     Qb     = (float*)(ws + OFF_Q);
  _Float16*  Bsw    = (_Float16*)(ws + OFF_BSW);
  float*     klp    = (float*)(ws + OFF_KL);

  prep_a<<<32, 256, 0, stream>>>(x, w1, s_ws, bparts, Bsw);
  prep_b<<<1, 256, 0, stream>>>(fc_w, fc_b, gam, bet, bparts, Pb, Qb);
  main_c<<<NROW/64, 128, 0, stream>>>(s_ws, Pb, Qb, Bsw, b1, w2, b2, noise, out, klp);
  reduce_d<<<1, 256, 0, stream>>>(klp, out);
}
